// GraphAttentionLayer_8246337208821
// MI455X (gfx1250) — compile-verified, matrix operand order checked by reading, not run on hardware
//
#include <hip/hip_runtime.h>
#include <stdint.h>

// GAT layer for MI455X (gfx1250, wave32).
// Flash-style fused masked-softmax + P@Wh with v_wmma_f32_16x16x32_bf16.
// HBM-bound on the 256MB adj stream (~11us @ 23.3TB/s); WMMA compute ~3-4us.

#define N_NODES 8192
#define IN_F    512
#define OUT_F   64
#define ALPHA   0.2f
#define PSTR    33    // dwords per p-tile row (32 + 1 pad): conflict-free A reads

typedef __attribute__((ext_vector_type(16))) __bf16 v16bf;
typedef __attribute__((ext_vector_type(8)))  float  v8f;

__device__ inline unsigned short f2bf(float f) {           // fp32 -> bf16 (RN)
  union { float f; uint32_t u; } x{f};
  return (unsigned short)((x.u + 0x8000u) >> 16);
}
// pack two fp32 -> two bf16 in one dword (lo = first), round-to-nearest
__device__ inline uint32_t pack2bf(float a, float b) {
  union { float f; uint32_t u; } xa{a}, xb{b};
  return ((xa.u + 0x8000u) >> 16) | ((xb.u + 0x8000u) & 0xFFFF0000u);
}

// ---------- Kernel A: Wh = h @ W (fp32 exact), plus bf16 transposed copy ----------
__global__ void __launch_bounds__(256) gat_wh(const float* __restrict__ h,
                                              const float* __restrict__ W,
                                              float* __restrict__ Wh,
                                              unsigned short* __restrict__ WhBT) {
  const int t = threadIdx.x;
  const int o = t & 63;
  const int r = blockIdx.x * 4 + (t >> 6);
  const float4* hr4 = (const float4*)(h + (size_t)r * IN_F);
  float acc = 0.f;
#pragma unroll 4
  for (int k = 0; k < IN_F / 4; ++k) {
    const float4 hv = hr4[k];
    acc = fmaf(hv.x, W[(4 * k + 0) * OUT_F + o], acc);
    acc = fmaf(hv.y, W[(4 * k + 1) * OUT_F + o], acc);
    acc = fmaf(hv.z, W[(4 * k + 2) * OUT_F + o], acc);
    acc = fmaf(hv.w, W[(4 * k + 3) * OUT_F + o], acc);
  }
  Wh[(size_t)r * OUT_F + o] = acc;
  WhBT[(size_t)o * N_NODES + r] = f2bf(acc);   // n-major copy for B fragments
}

// ---------- Kernel B: Wh1 = Wh@a[:64], Wh2 = Wh@a[64:] (one wave per row) ----------
__global__ void __launch_bounds__(256) gat_att(const float* __restrict__ Wh,
                                               const float* __restrict__ a,
                                               float* __restrict__ Wh1,
                                               float* __restrict__ Wh2) {
  const int lane = threadIdx.x & 31;
  const int row  = blockIdx.x * 8 + (threadIdx.x >> 5);
  const float x0 = Wh[(size_t)row * OUT_F + lane];
  const float x1 = Wh[(size_t)row * OUT_F + 32 + lane];
  float s1 = x0 * a[lane]      + x1 * a[32 + lane];
  float s2 = x0 * a[64 + lane] + x1 * a[96 + lane];
#pragma unroll
  for (int off = 16; off > 0; off >>= 1) {
    s1 += __shfl_xor(s1, off, 32);
    s2 += __shfl_xor(s2, off, 32);
  }
  if (lane == 0) { Wh1[row] = s1; Wh2[row] = s2; }
}

// ---------- Kernel C: fused masked softmax + attention@Wh + ELU ----------
// Block = 16 output rows, 4 waves; wave w owns the 16x16 n-tile at n0=16w.
// K-chunk = 64 columns -> two bf16 WMMAs per iteration, double-buffered p tile,
// one barrier per chunk. Each lane owns columns (2*lane, 2*lane+1).
__global__ void __launch_bounds__(128) gat_flash(const int* __restrict__ adj,
                                                 const float* __restrict__ Wh1,
                                                 const float* __restrict__ Wh2,
                                                 const unsigned short* __restrict__ WhBT,
                                                 float* __restrict__ out) {
  __shared__ __align__(16) uint32_t p_lds[2][16 * PSTR];   // bf16 pairs, 2 bufs
  __shared__ float l_lds[16];

  const int t    = threadIdx.x;
  const int lane = t & 31;
  const int w    = t >> 5;                 // wave id 0..3
  const int i0   = blockIdx.x * 16;        // first output row of this block
  const int n0   = w * 16;                 // this wave's output-column tile
  const int m16  = lane & 15;
  const int hi   = (lane >> 4) & 1;        // lane half (ISA A/B striping)

  // Score rows handled by this thread: m = w + 4q (all 16 rows across 4 waves)
  float wh1v[4];
  const int2* arow[4];                     // adj row pointers, int2 per lane
#pragma unroll
  for (int q = 0; q < 4; ++q) {
    const int m = w + 4 * q;
    wh1v[q] = Wh1[i0 + m];
    arow[q] = (const int2*)(adj + (size_t)(i0 + m) * N_NODES);
  }
  const float2* wh2p = (const float2*)Wh2;

  v8f c = {0.f, 0.f, 0.f, 0.f, 0.f, 0.f, 0.f, 0.f};
  float lacc[4] = {0.f, 0.f, 0.f, 0.f};

  // B fragment base: WhBT[n0+m16][j + hi*8 (+16)] -> contiguous b128 per load
  const unsigned short* bbase = WhBT + (size_t)(n0 + m16) * N_NODES + hi * 8;

  for (int j0 = 0; j0 < N_NODES; j0 += 64) {
    const int buf = (j0 >> 6) & 1;
    const int jp  = (j0 >> 1) + lane;      // int2 / float2 element index
    const float2 w2 = wh2p[jp];
    // --- branchless masked weights: mask folded into score, exp(-1e5) == 0 ---
#pragma unroll
    for (int q = 0; q < 4; ++q) {
      const int2 av = arow[q][jp];         // coalesced b64
      float s0 = wh1v[q] + w2.x;
      float s1 = wh1v[q] + w2.y;
      s0 = fmaxf(s0, ALPHA * s0);          // leaky_relu = max(s, 0.2s)
      s1 = fmaxf(s1, ALPHA * s1);
      s0 = av.x > 0 ? s0 : -1.0e5f;
      s1 = av.y > 0 ? s1 : -1.0e5f;
      const float p0 = __expf(s0);
      const float p1 = __expf(s1);
      lacc[q] += p0 + p1;
      p_lds[buf][(w + 4 * q) * PSTR + lane] = pack2bf(p0, p1);
      // Speculative (ISA 10.5: bad addresses silently dropped) -> no guard,
      // no per-iteration scalar branch. Streams adj 2 chunks ahead.
      __builtin_prefetch(&arow[q][jp + 64], 0, 0);
    }
    __syncthreads();                       // single barrier (double buffering)

    // --- A fragments (16x32 bf16 each) from LDS per ISA layout ---
    union { v16bf v; uint32_t u[8]; } A0, A1;
    const uint32_t* pl = &p_lds[buf][m16 * PSTR];
#pragma unroll
    for (int vv = 0; vv < 4; ++vv) {
      const int b = hi * 4 + vv;
      A0.u[vv]     = pl[b];                // k =      hi*8 + 2vv (,+1)
      A0.u[4 + vv] = pl[8 + b];            // k = 16 + hi*8 + 2vv
      A1.u[vv]     = pl[16 + b];           // k = 32 + ...
      A1.u[4 + vv] = pl[24 + b];           // k = 48 + ...
    }
    // --- B fragments (32x16 bf16 each) from L2-resident WhBT ---
    union { v16bf v; uint4 u4[2]; } B0, B1;
    B0.u4[0] = *(const uint4*)(bbase + j0);
    B0.u4[1] = *(const uint4*)(bbase + j0 + 16);
    B1.u4[0] = *(const uint4*)(bbase + j0 + 32);
    B1.u4[1] = *(const uint4*)(bbase + j0 + 48);

    c = __builtin_amdgcn_wmma_f32_16x16x32_bf16(false, A0.v, false, B0.v,
                                                (short)0, c, false, false);
    c = __builtin_amdgcn_wmma_f32_16x16x32_bf16(false, A1.v, false, B1.v,
                                                (short)0, c, false, false);
  }

  // --- l reduction (per row) then normalize + ELU + store ---
#pragma unroll
  for (int q = 0; q < 4; ++q) {
    float s = lacc[q];
#pragma unroll
    for (int off = 16; off > 0; off >>= 1) s += __shfl_xor(s, off, 32);
    if (lane == 0) l_lds[w + 4 * q] = s;
  }
  __syncthreads();

#pragma unroll
  for (int vv = 0; vv < 8; ++vv) {
    const int m = vv + hi * 8;             // ISA C/D layout
    const float l = l_lds[m];
    float val = c[vv] / (l > 1e-30f ? l : 1e-30f);
    val = val > 0.f ? val : expm1f(val);   // ELU(alpha=1)
    out[(size_t)(i0 + m) * OUT_F + n0 + m16] = val;
  }
}

extern "C" void kernel_launch(void* const* d_in, const int* in_sizes, int n_in,
                              void* d_out, int out_size, void* d_ws, size_t ws_size,
                              hipStream_t stream) {
  (void)in_sizes; (void)n_in; (void)out_size; (void)ws_size;
  const float* h   = (const float*)d_in[0];
  const int*   adj = (const int*)d_in[1];
  const float* W   = (const float*)d_in[2];
  const float* a   = (const float*)d_in[3];
  float* out = (float*)d_out;

  char* ws = (char*)d_ws;
  float*          Wh   = (float*)ws;                                          // 2 MB
  unsigned short* WhBT = (unsigned short*)(ws + (size_t)N_NODES * OUT_F * 4); // 1 MB
  float*          Wh1  = (float*)(ws + (size_t)N_NODES * OUT_F * 6);          // 32 KB
  float*          Wh2  = Wh1 + N_NODES;                                       // 32 KB

  hipLaunchKernelGGL(gat_wh,    dim3(N_NODES / 4),  dim3(256), 0, stream, h, W, Wh, WhBT);
  hipLaunchKernelGGL(gat_att,   dim3(N_NODES / 8),  dim3(256), 0, stream, Wh, a, Wh1, Wh2);
  hipLaunchKernelGGL(gat_flash, dim3(N_NODES / 16), dim3(128), 0, stream, adj, Wh1, Wh2, WhBT, out);
}
